// TCLM_32950989095381
// MI455X (gfx1250) — matrix operand order repaired and checked by reading, not compiled, not run on hardware
//
#include <hip/hip_runtime.h>
#include <hip/hip_bf16.h>

// ---------------------------------------------------------------------------
// Transformer LM forward for gfx1250 (MI455X, wave32, WMMA).
// All GEMMs + attention use v_wmma_f32_16x16x32_f16 (f32 accumulate).
// GEMM uses double-buffered LDS so global loads of slab i+1 overlap the
// WMMA burst of slab i (one workgroup barrier per K-step).
// ---------------------------------------------------------------------------

#define V_   32000
#define T_   1024
#define D_   1024
#define N_   6
#define H_   16
#define DFF_ 4096
#define DK_  64
#define B_   2
#define M_   (B_ * T_)      // 2048 token rows

typedef _Float16 f16;
typedef _Float16 v16h __attribute__((ext_vector_type(16)));
typedef _Float16 v8h  __attribute__((ext_vector_type(8)));
typedef float    v8f  __attribute__((ext_vector_type(8)));

union V16U { v16h v; v8h h[2]; };

__device__ inline v8f wmma32(v16h a, v16h b, v8f c) {
    // 8 args: (neg_a, A, neg_b, B, c_mod, C, reuse_a, reuse_b)
    return __builtin_amdgcn_wmma_f32_16x16x32_f16(false, a, false, b,
                                                  (short)0, c, false, false);
}

// A-fragment (16x32 f16): lane half-wave picks K 0-15 (lo) / 16-31 (hi), contiguous
__device__ inline v16h ld_a_frag(const f16* row, int ka) {
    V16U u;
    u.h[0] = *(const v8h*)(row + 2 * ka);
    u.h[1] = *(const v8h*)(row + 2 * ka + 8);
    return u.v;
}
// B-fragment: 16 contiguous halves (only 16B alignment required)
__device__ inline v16h ld_b_frag(const f16* p) {
    V16U u;
    u.h[0] = *(const v8h*)(p);
    u.h[1] = *(const v8h*)(p + 8);
    return u.v;
}

// ---------------------------------------------------------------------------
// Embedding + sinusoidal positional encoding:  x = emb[idx]*sqrt(D) + PE
// ---------------------------------------------------------------------------
__global__ __launch_bounds__(256) void k_embed(const int* __restrict__ idx,
                                               const float* __restrict__ emb,
                                               float* __restrict__ x) {
    int row = blockIdx.x;              // b*T + t
    int t   = row % T_;
    int tok = idx[row];
    const float* e  = emb + (size_t)tok * D_;
    float*       xr = x   + (size_t)row * D_;
    for (int d = threadIdx.x; d < D_; d += 256) {
        float ang = (float)t * __expf((float)(d & ~1) * (-9.210340372f / (float)D_));
        float pe  = (d & 1) ? cosf(ang) : sinf(ang);
        xr[d] = e[d] * 32.0f + pe;     // sqrt(1024) = 32
    }
}

// ---------------------------------------------------------------------------
// LayerNorm (torch-style: std with ddof=1, denom std+eps), f32 in -> f16 out
// ---------------------------------------------------------------------------
__global__ __launch_bounds__(256) void k_layernorm(const float* __restrict__ x,
                                                   const float* __restrict__ ga,
                                                   const float* __restrict__ gb,
                                                   f16* __restrict__ out) {
    __shared__ float rs[8], rss[8];
    __shared__ float stat[2];
    int row = blockIdx.x;
    const float* xr = x + (size_t)row * D_;
    float s = 0.f, ss = 0.f;
    for (int d = threadIdx.x; d < D_; d += 256) { float v = xr[d]; s += v; ss += v * v; }
#pragma unroll
    for (int off = 16; off >= 1; off >>= 1) {
        s  += __shfl_xor(s,  off, 32);
        ss += __shfl_xor(ss, off, 32);
    }
    if ((threadIdx.x & 31) == 0) { rs[threadIdx.x >> 5] = s; rss[threadIdx.x >> 5] = ss; }
    __syncthreads();
    if (threadIdx.x == 0) {
        float S = 0.f, SS = 0.f;
        for (int i = 0; i < 8; ++i) { S += rs[i]; SS += rss[i]; }
        float mean = S / (float)D_;
        float var  = (SS - (float)D_ * mean * mean) / (float)(D_ - 1);
        float sd   = sqrtf(fmaxf(var, 0.f));
        stat[0] = mean; stat[1] = 1.f / (sd + 1e-6f);
    }
    __syncthreads();
    float mean = stat[0], inv = stat[1];
    for (int d = threadIdx.x; d < D_; d += 256)
        out[(size_t)row * D_ + d] = (f16)(ga[d] * (xr[d] - mean) * inv + gb[d]);
}

// ---------------------------------------------------------------------------
// Weight transpose + f32 -> f16:  Wt[n][k] = (f16) W[k][n]
// ---------------------------------------------------------------------------
__global__ void k_transpose(const float* __restrict__ W, f16* __restrict__ Wt,
                            int K, int Ncols) {
    __shared__ float tile[32][33];
    int k0 = blockIdx.x * 32, n0 = blockIdx.y * 32;
    int tx = threadIdx.x, ty = threadIdx.y;          // block (32,8)
#pragma unroll
    for (int j = 0; j < 4; ++j)
        tile[ty + 8 * j][tx] = W[(size_t)(k0 + ty + 8 * j) * Ncols + n0 + tx];
    __syncthreads();
#pragma unroll
    for (int j = 0; j < 4; ++j)
        Wt[(size_t)(n0 + ty + 8 * j) * K + k0 + tx] = (f16)tile[tx][ty + 8 * j];
}

// ---------------------------------------------------------------------------
// WMMA GEMM:  C[M,N] = A[M,K](f16, row-major) * Bt[N,K](f16, row-major)^T
// 256 threads = 8 waves; block tile 128x128, K-step 32, wave tile 32x64.
// Double-buffered LDS: global loads for slab i+1 issue before slab i's WMMAs.
// ---------------------------------------------------------------------------
#define GBM 128
#define GBN 128
#define GBK 32
#define LDA 40                 // padded LDS row stride (halves), 80B = 16B-aligned

#define MODE_F16        0      // out f16 = acc
#define MODE_F16_RELU   1      // out f16 = relu(acc + bias)
#define MODE_F32_BIAS   2      // out f32 = acc + bias
#define MODE_RESID      3      // out f32 += acc (+ bias)

__global__ __launch_bounds__(256) void k_gemm(const f16* __restrict__ A,
                                              const f16* __restrict__ Bt,
                                              const float* __restrict__ bias,
                                              void* __restrict__ out,
                                              int Ndim, int Kdim, int mode) {
    __shared__ __align__(16) f16 As[2][GBM * LDA];
    __shared__ __align__(16) f16 Bs[2][GBN * LDA];

    int tid = threadIdx.x;
    int lane = tid & 31, wv = tid >> 5;
    int wm = wv & 3, wn = wv >> 2;                   // 4 waves (M) x 2 waves (N)
    int hw = lane >> 4, ln = lane & 15, ka = hw * 8;
    int m0 = blockIdx.x * GBM, n0 = blockIdx.y * GBN;

    v8f acc[2][4];
#pragma unroll
    for (int mi = 0; mi < 2; ++mi)
#pragma unroll
        for (int ni = 0; ni < 4; ++ni)
            acc[mi][ni] = (v8f){0.f,0.f,0.f,0.f,0.f,0.f,0.f,0.f};

    int ldrow = tid >> 2;                            // 0..63 (two rows: +0, +64)
    int ldcol = (tid & 3) << 3;                      // 0,8,16,24

    const f16* Ar0 = A  + (size_t)(m0 + ldrow)      * Kdim + ldcol;
    const f16* Ar1 = A  + (size_t)(m0 + ldrow + 64) * Kdim + ldcol;
    const f16* Br0 = Bt + (size_t)(n0 + ldrow)      * Kdim + ldcol;
    const f16* Br1 = Bt + (size_t)(n0 + ldrow + 64) * Kdim + ldcol;
    int sa0 = ldrow * LDA + ldcol, sa1 = (ldrow + 64) * LDA + ldcol;

    // preload slab 0
    v8h ra0 = *(const v8h*)(Ar0);
    v8h ra1 = *(const v8h*)(Ar1);
    v8h rb0 = *(const v8h*)(Br0);
    v8h rb1 = *(const v8h*)(Br1);
    *(v8h*)&As[0][sa0] = ra0;  *(v8h*)&As[0][sa1] = ra1;
    *(v8h*)&Bs[0][sa0] = rb0;  *(v8h*)&Bs[0][sa1] = rb1;
    __syncthreads();

    int p = 0;
    for (int kb = 0; kb < Kdim; kb += GBK) {
        int kn = kb + GBK;
        bool more = kn < Kdim;
        if (more) {                                  // issue next slab's loads now
            ra0 = *(const v8h*)(Ar0 + kn);
            ra1 = *(const v8h*)(Ar1 + kn);
            rb0 = *(const v8h*)(Br0 + kn);
            rb1 = *(const v8h*)(Br1 + kn);
            __builtin_prefetch(Ar0 + kn + GBK, 0, 3);   // WGP-scope prefetch
            __builtin_prefetch(Br0 + kn + GBK, 0, 3);
        }

        // compute current slab from LDS buffer p (overlaps the loads above)
        const f16* as = As[p];
        const f16* bs = Bs[p];
        v16h af[2], bf[4];
#pragma unroll
        for (int mi = 0; mi < 2; ++mi)
            af[mi] = ld_a_frag(&as[(wm * 32 + mi * 16 + ln) * LDA], ka);
#pragma unroll
        for (int ni = 0; ni < 4; ++ni)
            bf[ni] = ld_b_frag(&bs[(wn * 64 + ni * 16 + ln) * LDA + hw * 16]);
#pragma unroll
        for (int mi = 0; mi < 2; ++mi)
#pragma unroll
            for (int ni = 0; ni < 4; ++ni)
                acc[mi][ni] = wmma32(af[mi], bf[ni], acc[mi][ni]);

        if (more) {                                  // commit next slab, flip
            int q = p ^ 1;
            *(v8h*)&As[q][sa0] = ra0;  *(v8h*)&As[q][sa1] = ra1;
            *(v8h*)&Bs[q][sa0] = rb0;  *(v8h*)&Bs[q][sa1] = rb1;
            __syncthreads();
            p = q;
        }
    }

    // epilogue: lane owns (m = r + 8*hw, n = ln) of each 16x16 tile
#pragma unroll
    for (int mi = 0; mi < 2; ++mi) {
#pragma unroll
        for (int ni = 0; ni < 4; ++ni) {
#pragma unroll
            for (int r = 0; r < 8; ++r) {
                int m = m0 + wm * 32 + mi * 16 + r + 8 * hw;
                int n = n0 + wn * 64 + ni * 16 + ln;
                size_t o = (size_t)m * Ndim + n;
                float v = acc[mi][ni][r];
                if (mode == MODE_F16) {
                    ((f16*)out)[o] = (f16)v;
                } else if (mode == MODE_F16_RELU) {
                    ((f16*)out)[o] = (f16)fmaxf(v + bias[n], 0.f);
                } else if (mode == MODE_F32_BIAS) {
                    ((float*)out)[o] = v + bias[n];
                } else {                             // MODE_RESID
                    float bb = bias ? bias[n] : 0.f;
                    ((float*)out)[o] += v + bb;
                }
            }
        }
    }
}

// ---------------------------------------------------------------------------
// Head repacks (f16) from fused QKV output [M, 3D]:
//   -> [B,H,T,DK]  (Q, K)   and   -> [B,H,DK,T]  (V^T)
// ---------------------------------------------------------------------------
__global__ __launch_bounds__(256) void k_repack_head(const f16* __restrict__ in,
                                                     f16* __restrict__ out,
                                                     int ldin, int coloff) {
    int i  = blockIdx.x * 256 + threadIdx.x;         // over B*T*D
    int dk = i & 63, h = (i >> 6) & 15, t = (i >> 10) & 1023, b = i >> 20;
    f16 v = in[(size_t)((b << 10) + t) * ldin + coloff + (h << 6) + dk];
    out[(((size_t)(b * H_ + h) * T_ + t) * DK_) + dk] = v;
}
__global__ __launch_bounds__(256) void k_repack_vt(const f16* __restrict__ in,
                                                   f16* __restrict__ out,
                                                   int ldin, int coloff) {
    int i  = blockIdx.x * 256 + threadIdx.x;
    int dk = i & 63, h = (i >> 6) & 15, t = (i >> 10) & 1023, b = i >> 20;
    f16 v = in[(size_t)((b << 10) + t) * ldin + coloff + (h << 6) + dk];
    out[(((size_t)(b * H_ + h) * DK_ + dk) * T_) + t] = v;
}

// ---------------------------------------------------------------------------
// Flash attention, causal. One wave per 16-row query tile (4 waves / block).
// S = Q K^T / 8 via WMMA; online softmax (rows shared by 16 lanes -> shfl_xor
// reductions within each half-wave); P routed through per-wave LDS to convert
// C-layout -> A-layout; O += P V via WMMA against pre-transposed V^T.
// ---------------------------------------------------------------------------
__global__ __launch_bounds__(128) void k_attn(const f16* __restrict__ qh,
                                              const f16* __restrict__ kh,
                                              const f16* __restrict__ vt,
                                              f16* __restrict__ o16) {
    __shared__ __align__(16) f16 pl[4][16][40];      // per-wave P patch (16x32 +pad)

    int lane = threadIdx.x & 31, wid = threadIdx.x >> 5;
    int gw = blockIdx.x * 4 + wid;                   // global q-tile id
    int qt = gw & 63;                                // T/16 = 64 tiles
    int h  = (gw >> 6) & (H_ - 1);
    int b  = gw >> 10;
    int hw = lane >> 4, ln = lane & 15, ka = hw * 8;
    size_t bh = (size_t)(b * H_ + h);

    // Q fragments (rows qt*16+ln, features 0..31 and 32..63)
    const f16* Qr = qh + (bh * T_ + qt * 16 + ln) * DK_;
    v16h qa0 = ld_a_frag(Qr, ka);
    v16h qa1 = ld_a_frag(Qr + 32, ka);

    v8f accO[4];
#pragma unroll
    for (int nt = 0; nt < 4; ++nt)
        accO[nt] = (v8f){0.f,0.f,0.f,0.f,0.f,0.f,0.f,0.f};
    float mrow[8], lrow[8];
#pragma unroll
    for (int r = 0; r < 8; ++r) { mrow[r] = -1e30f; lrow[r] = 0.f; }

    int kend = qt * 16 + 15;                         // causal upper key bound
    for (int kt = 0; kt <= kend; kt += 32) {
        const f16* K0 = kh + (bh * T_ + kt + ln) * DK_;        // keys kt..kt+15
        const f16* K1 = kh + (bh * T_ + kt + 16 + ln) * DK_;   // keys kt+16..kt+31
        v8f s0 = (v8f){0.f,0.f,0.f,0.f,0.f,0.f,0.f,0.f};
        v8f s1 = s0;
        s0 = wmma32(qa0, ld_b_frag(K0 + hw * 16), s0);
        s0 = wmma32(qa1, ld_b_frag(K0 + 32 + hw * 16), s0);
        s1 = wmma32(qa0, ld_b_frag(K1 + hw * 16), s1);
        s1 = wmma32(qa1, ld_b_frag(K1 + 32 + hw * 16), s1);

#pragma unroll
        for (int r = 0; r < 8; ++r) {
            int q    = qt * 16 + r + 8 * hw;
            int key0 = kt + ln, key1 = kt + 16 + ln;
            float v0 = (key0 > q) ? -1e30f : s0[r] * 0.125f;   // 1/sqrt(64)
            float v1 = (key1 > q) ? -1e30f : s1[r] * 0.125f;
            float mx = fmaxf(v0, v1);
#pragma unroll
            for (int off = 8; off >= 1; off >>= 1)
                mx = fmaxf(mx, __shfl_xor(mx, off, 32));
            float mnew = fmaxf(mrow[r], mx);
            float al   = __expf(mrow[r] - mnew);
            float p0   = __expf(v0 - mnew);
            float p1   = __expf(v1 - mnew);
            float rsum = p0 + p1;
#pragma unroll
            for (int off = 8; off >= 1; off >>= 1)
                rsum += __shfl_xor(rsum, off, 32);
            lrow[r] = lrow[r] * al + rsum;
            mrow[r] = mnew;
#pragma unroll
            for (int nt = 0; nt < 4; ++nt) accO[nt][r] *= al;
            int prow = r + 8 * hw;                  // C-layout -> LDS
            pl[wid][prow][ln]      = (f16)p0;
            pl[wid][prow][16 + ln] = (f16)p1;
        }

        // P back as A-fragment (row = ln), then O += P * V
        v16h pa = ld_a_frag(&pl[wid][ln][0], ka);
#pragma unroll
        for (int nt = 0; nt < 4; ++nt) {
            const f16* Vb = vt + ((bh * DK_) + nt * 16 + ln) * T_ + kt + hw * 16;
            accO[nt] = wmma32(pa, ld_b_frag(Vb), accO[nt]);
        }
    }

    // O / l -> [B,T,D] f16
#pragma unroll
    for (int nt = 0; nt < 4; ++nt) {
#pragma unroll
        for (int r = 0; r < 8; ++r) {
            int tq = qt * 16 + r + 8 * hw;
            o16[((size_t)(b * T_ + tq)) * D_ + h * DK_ + nt * 16 + ln] =
                (f16)(accO[nt][r] / lrow[r]);
        }
    }
}

// ---------------------------------------------------------------------------
// Host orchestration (all on `stream`, capture-safe: launches only)
// ---------------------------------------------------------------------------
extern "C" void kernel_launch(void* const* d_in, const int* in_sizes, int n_in,
                              void* d_out, int out_size, void* d_ws, size_t ws_size,
                              hipStream_t stream) {
    (void)in_sizes; (void)n_in; (void)out_size; (void)ws_size;
    const int*   idx  = (const int*)  d_in[0];
    const float* emb  = (const float*)d_in[1];
    const float* Wq   = (const float*)d_in[2];
    const float* Wk   = (const float*)d_in[3];
    const float* Wv   = (const float*)d_in[4];
    const float* Wo   = (const float*)d_in[5];
    const float* ln1a = (const float*)d_in[6];
    const float* ln1b = (const float*)d_in[7];
    const float* ln2a = (const float*)d_in[8];
    const float* ln2b = (const float*)d_in[9];
    const float* W1   = (const float*)d_in[10];
    const float* b1   = (const float*)d_in[11];
    const float* W2   = (const float*)d_in[12];
    const float* b2   = (const float*)d_in[13];
    const float* fna  = (const float*)d_in[14];
    const float* fnb  = (const float*)d_in[15];
    const float* Wp   = (const float*)d_in[16];
    const float* bp   = (const float*)d_in[17];
    float* logits = (float*)d_out;

    char* w = (char*)d_ws;
    auto take = [&](size_t bytes) { char* p = w; w += (bytes + 255) & ~(size_t)255; return p; };
    float* x    = (float*)take((size_t)M_ * D_ * 4);
    f16* hln    = (f16*)take((size_t)M_ * D_ * 2);
    f16* qkv16  = (f16*)take((size_t)M_ * 3 * D_ * 2);   // fused QKV out [M, 3D]
    f16* qhb    = (f16*)take((size_t)M_ * D_ * 2);
    f16* khb    = (f16*)take((size_t)M_ * D_ * 2);
    f16* vtb    = (f16*)take((size_t)M_ * D_ * 2);
    f16* att    = (f16*)take((size_t)M_ * D_ * 2);
    f16* ff1    = (f16*)take((size_t)M_ * DFF_ * 2);
    f16* wtqkv  = (f16*)take((size_t)3 * D_ * D_ * 2);   // [3D, D] = Wq^T|Wk^T|Wv^T
    f16* wt0    = (f16*)take((size_t)D_ * D_ * 2);       // Wo^T
    f16* wt1    = (f16*)take((size_t)D_ * DFF_ * 2);
    f16* wt2    = (f16*)take((size_t)DFF_ * D_ * 2);
    f16* wtp    = (f16*)take((size_t)D_ * V_ * 2);

    const dim3 tb(32, 8);
    const dim3 gDD(D_ / 32, D_ / 32);
    const int  rp = (B_ * T_ * D_) / 256;                // repack blocks

    k_embed<<<M_, 256, 0, stream>>>(idx, emb, x);

    for (int l = 0; l < N_; ++l) {
        size_t oDD = (size_t)l * D_ * D_;
        k_layernorm<<<M_, 256, 0, stream>>>(x, ln1a + l * D_, ln1b + l * D_, hln);

        // fused QKV projection: one GEMM, N = 3072
        k_transpose<<<gDD, tb, 0, stream>>>(Wq + oDD, wtqkv,                     D_, D_);
        k_transpose<<<gDD, tb, 0, stream>>>(Wk + oDD, wtqkv + (size_t)D_ * D_,   D_, D_);
        k_transpose<<<gDD, tb, 0, stream>>>(Wv + oDD, wtqkv + (size_t)2*D_*D_,   D_, D_);
        k_gemm<<<dim3(M_ / 128, (3 * D_) / 128), 256, 0, stream>>>(
            hln, wtqkv, nullptr, qkv16, 3 * D_, D_, MODE_F16);

        k_repack_head<<<rp, 256, 0, stream>>>(qkv16, qhb, 3 * D_, 0);
        k_repack_head<<<rp, 256, 0, stream>>>(qkv16, khb, 3 * D_, D_);
        k_repack_vt  <<<rp, 256, 0, stream>>>(qkv16, vtb, 3 * D_, 2 * D_);

        k_attn<<<(B_ * H_ * (T_ / 16)) / 4, 128, 0, stream>>>(qhb, khb, vtb, att);

        k_transpose<<<gDD, tb, 0, stream>>>(Wo + oDD, wt0, D_, D_);
        k_gemm<<<dim3(M_ / 128, D_ / 128), 256, 0, stream>>>(att, wt0, nullptr, x, D_, D_, MODE_RESID);

        k_layernorm<<<M_, 256, 0, stream>>>(x, ln2a + l * D_, ln2b + l * D_, hln);

        k_transpose<<<dim3(D_ / 32, DFF_ / 32), tb, 0, stream>>>(W1 + (size_t)l * D_ * DFF_, wt1, D_, DFF_);
        k_gemm<<<dim3(M_ / 128, DFF_ / 128), 256, 0, stream>>>(hln, wt1, b1 + l * DFF_, ff1, DFF_, D_, MODE_F16_RELU);
        k_transpose<<<dim3(DFF_ / 32, D_ / 32), tb, 0, stream>>>(W2 + (size_t)l * DFF_ * D_, wt2, DFF_, D_);
        k_gemm<<<dim3(M_ / 128, D_ / 128), 256, 0, stream>>>(ff1, wt2, b2 + l * D_, x, D_, DFF_, MODE_RESID);
    }

    k_layernorm<<<M_, 256, 0, stream>>>(x, fna, fnb, hln);
    k_transpose<<<dim3(D_ / 32, V_ / 32), tb, 0, stream>>>(Wp, wtp, D_, V_);
    k_gemm<<<dim3(M_ / 128, V_ / 128), 256, 0, stream>>>(hln, wtp, bp, logits, V_, D_, MODE_F32_BIAS);
}
